// SSMSynapticDelay_70489003262049
// MI455X (gfx1250) — matrix operand order repaired and checked by reading, not compiled, not run on hardware
//
#include <hip/hip_runtime.h>
#include <hip/hip_bf16.h>
#include <math.h>

// Problem constants (from reference): B=2048, S=200, C=25, H=8
#define SREAL 200
#define CREAL 25
#define HDIM  8
#define SP    208   // S padded to 13 WMMA tiles of 16
#define XROWS 224   // t-rows padded to 7 chunks of 32 (K dim of x_del GEMM)
#define CP    32    // C padded to one K=32 slab

typedef __attribute__((ext_vector_type(16))) _Float16 v16h;
typedef __attribute__((ext_vector_type(8)))  float    v8f;
typedef int v4i __attribute__((vector_size(4 * sizeof(int))));

// Async-to-LDS builtin takes (AS1 v4i* gsrc, AS3 v4i* ldst, imm offset, imm cpol)
#define AS_GLOBAL_V4(p) ((__attribute__((address_space(1))) v4i*)(p))
#define AS_LOCAL_V4(p)  ((__attribute__((address_space(3))) v4i*)(p))

// ---- WMMA fragment loaders from LDS, contiguous k order (k = 16*(L/16) + j) for A and B ----
// A (16x32 f16): lane L holds row m=L%16; halves j=0..15 -> K=kb+j,
// kb = (L<16)?0:16.  Row-major storage: contiguous 32B run -> ds_load_b128.
__device__ __forceinline__ v16h lds_load_a_rowmajor(const _Float16* base, int stride,
                                                    int mBase, int lane) {
  const int m  = mBase + (lane & 15);
  const int kb = (lane < 16) ? 0 : 16;
  v16h a;
#pragma unroll
  for (int j = 0; j < 16; ++j)
    a[j] = base[m * stride + kb + j];
  return a;
}

// B (32x16 f16): lane L holds column n=L%16; halves j=0..15 -> K=kb+j, kb=(L<16)?0:16.
// Storage with the B-column contiguous: element (k,n) = base[(nBase+n)*stride + kRow0 + k]
// -> one contiguous 32B run per lane -> 2x ds_load_b128.
__device__ __forceinline__ v16h lds_load_b_ncontig(const _Float16* base, int stride,
                                                   int nBase, int kRow0, int lane) {
  const int n  = nBase + (lane & 15);
  const int kb = (lane < 16) ? 0 : 16;
  v16h b;
#pragma unroll
  for (int j = 0; j < 16; ++j)
    b[j] = base[n * stride + kRow0 + kb + j];
  return b;
}

__global__ __launch_bounds__(256)
void ssm_synaptic_delay_fused(const float* __restrict__ x,        // [B,S,C]
                              const float* __restrict__ dW,       // [S,C]
                              const float* __restrict__ db,       // [S]
                              const float* __restrict__ UW,       // [H,C]
                              const float* __restrict__ Ub,       // [H]
                              const float* __restrict__ AW,       // [H,H]
                              const float* __restrict__ BW,       // [H,C]
                              const float* __restrict__ Bb,       // [H]
                              const float* __restrict__ oW,       // [C,H]
                              const float* __restrict__ ob,       // [C]
                              float* __restrict__ out) {          // [B,S,C]
  // One workgroup per batch element; all intermediates live in the 320KB WGP LDS.
  __shared__ _Float16 Xh [XROWS * CP];  // x_b  [t][c] f16                 14.0 KB
  __shared__ _Float16 XhT[CP * XROWS];  // x_b  [c][t] f16 (B-frag layout) 14.0 KB
  __shared__ _Float16 DWh[SP * CP];     // delay_W [s][c] f16              13.0 KB
  __shared__ float    dbl[SP];          // delay_b padded                   0.8 KB
  __shared__ float    Lt[32 * SP];      // logits chunk [32 t][208 s]      26.0 KB  (also dW f32 stage)
  __shared__ _Float16 WhT[SP * 32];     // softmax weights [s][t] f16      13.0 KB
  __shared__ float    XD[SP * CP];      // x_del f32                       26.0 KB  (also x f32 stage)
  __shared__ float    BU[SP * HDIM];    // b_t * u_t                        6.5 KB
  __shared__ float    HS[SP * HDIM];    // scan outputs h_s                 6.5 KB
  __shared__ float    Uw[HDIM * CP], Bw[HDIM * CP], Aw[HDIM * HDIM];
  __shared__ float    Ow[CREAL * HDIM], Ubv[HDIM], Bbv[HDIM], Obv[CREAL];

  const int tid  = threadIdx.x;
  const int wave = tid >> 5;
  const int lane = tid & 31;
  const int b    = blockIdx.x;

  // ------- Phase 0a: stage x_b and delay_W into LDS f32 staging (reuse XD / Lt) -----
  // Async DMA (ASYNCcnt-tracked) when available; 5000 f32 each = 1250 b128 beats.
  const float* xb = x + (size_t)b * (SREAL * CREAL);
#if __has_builtin(__builtin_amdgcn_global_load_async_to_lds_b128)
  for (int i = tid; i < 1250; i += 256)
    __builtin_amdgcn_global_load_async_to_lds_b128(AS_GLOBAL_V4(xb + i * 4),
                                                   AS_LOCAL_V4(&XD[i * 4]), 0, 0);
  for (int i = tid; i < 1250; i += 256)
    __builtin_amdgcn_global_load_async_to_lds_b128(AS_GLOBAL_V4(dW + i * 4),
                                                   AS_LOCAL_V4(&Lt[i * 4]), 0, 0);
#else
  for (int i = tid; i < SREAL * CREAL; i += 256) XD[i] = xb[i];
  for (int i = tid; i < SREAL * CREAL; i += 256) Lt[i] = dW[i];
#endif

  // Small weights via normal loads while the DMA is in flight.
  for (int i = tid; i < SP; i += 256) dbl[i] = (i < SREAL) ? db[i] : 0.f;
  for (int i = tid; i < HDIM * CP; i += 256) {
    const int h = i >> 5, c = i & 31;
    Uw[i] = (c < CREAL) ? UW[h * CREAL + c] : 0.f;
    Bw[i] = (c < CREAL) ? BW[h * CREAL + c] : 0.f;
  }
  if (tid < HDIM * HDIM) Aw[tid] = AW[tid];
  for (int i = tid; i < CREAL * HDIM; i += 256) Ow[i] = oW[i];
  if (tid < HDIM) { Ubv[tid] = Ub[tid]; Bbv[tid] = Bb[tid]; }
  if (tid < CREAL) Obv[tid] = ob[tid];

#if __has_builtin(__builtin_amdgcn_s_wait_asynccnt)
  __builtin_amdgcn_s_wait_asynccnt(0);
#else
  asm volatile("s_wait_asynccnt 0" ::: "memory");
#endif
  __syncthreads();

  // ------- Phase 0b: convert staged f32 -> f16 operand layouts (zero padded) --------
  for (int i = tid; i < XROWS * CP; i += 256) {
    const int t = i >> 5, c = i & 31;
    const _Float16 v = (_Float16)((t < SREAL && c < CREAL) ? XD[t * CREAL + c] : 0.f);
    Xh[t * CP + c]     = v;
    XhT[c * XROWS + t] = v;
  }
  for (int i = tid; i < SP * CP; i += 256) {
    const int s = i >> 5, c = i & 31;
    DWh[i] = (_Float16)((s < SREAL && c < CREAL) ? Lt[s * CREAL + c] : 0.f);
  }
  __syncthreads();

  // x_del accumulators: 26 output tiles (13 s-tiles x 2 c-tiles) over 8 waves -> <=4 each
  v8f accXD[4];
#pragma unroll
  for (int s4 = 0; s4 < 4; ++s4)
#pragma unroll
    for (int r = 0; r < 8; ++r) accXD[s4][r] = 0.f;

  // ---------------- Main loop over 7 chunks of 32 source timesteps t -----------------
  for (int tc = 0; tc < 7; ++tc) {
    const int t0 = tc * 32;

    // Phase 1a: logits tile  L[t_local, s] = (X · delay_W^T)[t, s] + delay_b[s]
    for (int job = wave; job < 26; job += 8) {        // wave-uniform: EXEC stays all-1s
      const int rt = job / 13, sj = job % 13;
      const v16h a  = lds_load_a_rowmajor(Xh, CP, t0 + rt * 16, lane);
      const v16h bm = lds_load_b_ncontig(DWh, CP, sj * 16, 0, lane);
      v8f c;
#pragma unroll
      for (int r = 0; r < 8; ++r) c[r] = 0.f;
      c = __builtin_amdgcn_wmma_f32_16x16x32_f16(false, a, false, bm,
                                                 (short)0, c, false, false);
      const int g = lane >> 4, n = lane & 15;
#pragma unroll
      for (int r = 0; r < 8; ++r)
        Lt[(rt * 16 + r + 8 * g) * SP + sj * 16 + n] = c[r] + dbl[sj * 16 + n];
    }
    __syncthreads();

    // Phase 1b: row softmax over s (16-lane shuffle reductions), emit f16 weights
    // transposed: WhT[s][t_local] so the phase-1c A-fragment is contiguous.
    {
      const int grp = tid >> 4, l16 = tid & 15;
#pragma unroll
      for (int rr = 0; rr < 2; ++rr) {
        const int row = grp + rr * 16;
        const int tg  = t0 + row;
        float e[13];
        float mx = -3.0e38f;
#pragma unroll
        for (int i = 0; i < 13; ++i) {
          const int s = l16 + i * 16;
          const float v = (s < SREAL) ? Lt[row * SP + s] : -3.0e38f;
          e[i] = v;
          mx = fmaxf(mx, v);
        }
#pragma unroll
        for (int off = 8; off; off >>= 1) mx = fmaxf(mx, __shfl_xor(mx, off, 16));
        float sum = 0.f;
#pragma unroll
        for (int i = 0; i < 13; ++i) {
          const int s = l16 + i * 16;
          const float ev = (s < SREAL) ? __expf(e[i] - mx) : 0.f;
          e[i] = ev;
          sum += ev;
        }
#pragma unroll
        for (int off = 8; off; off >>= 1) sum += __shfl_xor(sum, off, 16);
        const float inv = (tg < SREAL) ? (1.f / sum) : 0.f;  // kill padded t rows
#pragma unroll
        for (int i = 0; i < 13; ++i)
          WhT[(l16 + i * 16) * 32 + row] = (_Float16)(e[i] * inv);
      }
    }
    __syncthreads();

    // Phase 1c: accumulate x_del[s,c] += (w^T · X) over this K=32 t-slab.
    // A = w^T: element (m=s,k=t) = WhT[s][t]  (row-major, contiguous)
    // B = X  : element (k=t,n=c) = XhT[c][t0+t] (column-contiguous)
#pragma unroll
    for (int slot = 0; slot < 4; ++slot) {
      const int tile = wave + slot * 8;
      if (tile < 26) {                                // wave-uniform branch
        const int sm = tile % 13, cn = tile / 13;
        const v16h a  = lds_load_a_rowmajor(WhT, 32, sm * 16, lane);
        const v16h bm = lds_load_b_ncontig(XhT, XROWS, cn * 16, t0, lane);
        accXD[slot] = __builtin_amdgcn_wmma_f32_16x16x32_f16(false, a, false, bm,
                                                             (short)0, accXD[slot],
                                                             false, false);
      }
    }
    __syncthreads();
  }

  // ---------------- Phase 2: spill x_del accumulators to LDS -------------------------
#pragma unroll
  for (int slot = 0; slot < 4; ++slot) {
    const int tile = wave + slot * 8;
    if (tile < 26) {
      const int sm = tile % 13, cn = tile / 13;
      const int g = lane >> 4, n = lane & 15;
#pragma unroll
      for (int r = 0; r < 8; ++r)
        XD[(sm * 16 + r + 8 * g) * CP + cn * 16 + n] = accXD[slot][r];
    }
  }
  __syncthreads();

  // ---------------- Phase 3: u = XD·U_W^T + U_b ; bu = sigmoid(XD·B_W^T + B_b) * u ---
  for (int i = tid; i < SP * HDIM; i += 256) {
    const int s = i >> 3, h = i & 7;
    float us = Ubv[h], bs = Bbv[h];
#pragma unroll
    for (int c = 0; c < CREAL; ++c) {
      const float xv = XD[s * CP + c];
      us = fmaf(xv, Uw[h * CP + c], us);
      bs = fmaf(xv, Bw[h * CP + c], bs);
    }
    BU[i] = us * (1.f / (1.f + __expf(-bs)));
  }
  __syncthreads();

  // ---------------- Phase 4: linear scan h_s = h_{s-1}·A^T + bu_s (wave 0, H=8) ------
  if (tid < 32) {
    const int i = lane & 7;                 // lanes 8..31 mirror lanes 0..7
    float ar[HDIM];
#pragma unroll
    for (int j = 0; j < HDIM; ++j) ar[j] = Aw[i * HDIM + j];
    float h = 0.f;
    for (int s = 0; s < SREAL; ++s) {
      float hn = 0.f;
#pragma unroll
      for (int j = 0; j < HDIM; ++j)
        hn = fmaf(ar[j], __shfl(h, j, 32), hn);
      h = hn + BU[s * HDIM + i];
      if (lane < HDIM) HS[s * HDIM + i] = h;
    }
  }
  __syncthreads();

  // ---------------- Phase 5: out = HS·out_W^T + out_b, streamed to HBM ---------------
  float* outb = out + (size_t)b * (SREAL * CREAL);
  for (int i = tid; i < SREAL * CREAL; i += 256) {
    const int s = i / CREAL, c = i % CREAL;
    float acc = Obv[c];
#pragma unroll
    for (int h = 0; h < HDIM; ++h)
      acc = fmaf(HS[s * HDIM + h], Ow[c * HDIM + h], acc);
    outb[i] = acc;
  }
}

extern "C" void kernel_launch(void* const* d_in, const int* in_sizes, int n_in,
                              void* d_out, int out_size, void* d_ws, size_t ws_size,
                              hipStream_t stream) {
  const float* x  = (const float*)d_in[0];
  const float* dW = (const float*)d_in[1];
  const float* db = (const float*)d_in[2];
  const float* UW = (const float*)d_in[3];
  const float* Ub = (const float*)d_in[4];
  const float* AW = (const float*)d_in[5];
  const float* BW = (const float*)d_in[6];
  const float* Bb = (const float*)d_in[7];
  const float* oW = (const float*)d_in[8];
  const float* ob = (const float*)d_in[9];
  float* out = (float*)d_out;

  const int Bsz = in_sizes[0] / (SREAL * CREAL);   // 2048
  ssm_synaptic_delay_fused<<<dim3(Bsz), dim3(256), 0, stream>>>(
      x, dW, db, UW, Ub, AW, BW, Bb, oW, ob, out);
}
